// TimesFMAttention_86560770883652
// MI455X (gfx1250) — matrix operand order repaired and checked by reading, not compiled, not run on hardware
//
#include <hip/hip_runtime.h>
#include <hip/hip_bf16.h>
#include <math.h>

// ---------------------------------------------------------------------------
// TimesFM attention for MI455X (gfx1250, wave32, WMMA bf16 + async-LDS path)
// ---------------------------------------------------------------------------
typedef __attribute__((ext_vector_type(16))) __bf16 v16bf;
typedef __attribute__((ext_vector_type(8)))  float  v8f;

union Frag16 {
    v16bf          v;
    unsigned       u[8];
    unsigned short us[16];
};

__device__ __forceinline__ unsigned short f2bf(float f) {
    unsigned x = __builtin_bit_cast(unsigned, f);
    unsigned r = x + 0x7FFFu + ((x >> 16) & 1u);   // round-to-nearest-even
    return (unsigned short)(r >> 16);
}

// 16-bit A-matrix 16x32 layout (contiguous, matches B): VGPR v holds K-pair (2v,2v+1)
//   v0:(0,1) v1:(2,3) ... v7:(14,15)
// lanes 16..31 shift K by +16.
__device__ __forceinline__ int a_kbase(int v, int lane) {
    int b = v << 1;
    return b + ((lane & 16) ? 16 : 0);
}

#define WMMA_BF16(acc, a, b)                                                   \
    acc = __builtin_amdgcn_wmma_f32_16x16x32_bf16(false, (a), false, (b),      \
                                                  (short)0, (acc), false, false)

// Async global->LDS 16-byte copy (CDNA5 GLOBAL_LOAD_ASYNC_TO_LDS_B128).
// lds_off: byte offset within the workgroup's LDS allocation (low 32 bits of
// a generic LDS pointer). gptr: 64-bit global address. Tracked by ASYNCcnt.
__device__ __forceinline__ void async_copy_b128(unsigned lds_off, const void* gptr) {
    asm volatile("global_load_async_to_lds_b128 %0, %1, off"
                 :: "v"(lds_off), "v"(gptr)
                 : "memory");
}
__device__ __forceinline__ void async_wait0() {
    asm volatile("s_wait_asynccnt 0" ::: "memory");
}
__device__ __forceinline__ unsigned lds_off32(const void* p) {
    return (unsigned)(unsigned long long)p;   // LDS offset lives in low 32 bits
}

// Problem constants
constexpr int   Bc    = 2;
constexpr int   Sc    = 2048;
constexpr int   Hc    = 1024;
constexpr int   NHc   = 16;
constexpr int   NTOK  = Bc * Sc;            // 4096
constexpr float NEGF  = -2.3819763e+38f;
constexpr float RSP   = 1.442695041f;

// ---------------------------------------------------------------------------
// Kernel 1: QKV projection.  Y = X @ W^T + b, split/route to Q (scaled), K, V
// Tile: 128(M) x 64(N) per 256-thread block, 8 waves, wave tile 16x64.
// (fp32 -> bf16 conversion forces the VALU staging path here.)
// ---------------------------------------------------------------------------
__global__ __launch_bounds__(256)
void qkv_gemm(const float* __restrict__ X, const float* __restrict__ W,
              const float* __restrict__ bias, const float* __restrict__ scaling,
              unsigned short* __restrict__ Qb, unsigned short* __restrict__ Kb,
              unsigned short* __restrict__ Vb)
{
    __shared__ alignas(32) unsigned short As[128 * 32];
    __shared__ alignas(32) unsigned short Bs[64 * 32];

    const int tid  = threadIdx.x;
    const int lane = tid & 31;
    const int w    = tid >> 5;
    const int ntile = blockIdx.x % 24;       // 1536 / 64
    const int mtile = blockIdx.x / 24;       // 4096 / 128
    const int m0 = mtile * 128, n0 = ntile * 64;
    const int m    = lane & 15;
    const int half = lane & 16;

    v8f acc[4] = {};

    for (int kk = 0; kk < Hc; kk += 32) {
        __syncthreads();
        // stage A (fp32 -> bf16): 128x32
        #pragma unroll
        for (int i = 0; i < 16; ++i) {
            int e = tid + 256 * i;
            int r = e >> 5, c = e & 31;
            As[e] = f2bf(X[(size_t)(m0 + r) * Hc + kk + c]);
        }
        // stage B (W row-major 1536x1024): Bs[n][k] = W[n0+n][kk+k]
        #pragma unroll
        for (int i = 0; i < 8; ++i) {
            int e = tid + 256 * i;
            int r = e >> 5, c = e & 31;
            Bs[e] = f2bf(W[(size_t)(n0 + r) * Hc + kk + c]);
        }
        if (kk + 32 < Hc)
            __builtin_prefetch(&X[(size_t)(m0 + (tid & 127)) * Hc + kk + 32], 0, 1);
        __syncthreads();

        Frag16 a;
        #pragma unroll
        for (int v = 0; v < 8; ++v)
            a.u[v] = *(const unsigned*)&As[(16 * w + m) * 32 + a_kbase(v, lane)];

        #pragma unroll
        for (int c = 0; c < 4; ++c) {
            Frag16 b;
            b.v = *(const v16bf*)&Bs[(c * 16 + m) * 32 + half];
            WMMA_BF16(acc[c], a.v, b.v);
        }
    }

    // Epilogue: bias, Q-scale (R/sqrt(HD) * softplus), route to Q/K/V (bf16)
    const int mrow = m0 + 16 * w + (half ? 8 : 0);
    #pragma unroll
    for (int c = 0; c < 4; ++c) {
        const int gn  = n0 + c * 16 + m;
        const float bb = bias[gn];
        float sc = 1.0f;
        if (gn < 1024) sc = (RSP / 8.0f) * log1pf(__expf(scaling[gn & 63]));
        #pragma unroll
        for (int j = 0; j < 8; ++j) {
            const int gm = mrow + j;
            float val = acc[c][j] + bb;
            if (gn < 1024) {
                Qb[(size_t)gm * 1024 + gn] = f2bf(val * sc);
            } else if (gn < 1280) {
                Kb[(size_t)gm * 256 + (gn - 1024)] = f2bf(val);
            } else {
                Vb[(size_t)gm * 256 + (gn - 1280)] = f2bf(val);
            }
        }
    }
}

// ---------------------------------------------------------------------------
// Kernel 2: scores = softmax(Q K^T + causal mask), written fp32 to d_out.
// One block per (b, h, 16 q-rows). 8 waves; wave handles 16 key-tiles of 16.
// Q tile is async-copied global->LDS once (pure bf16 copy) instead of each
// wave re-gathering it from global.
// ---------------------------------------------------------------------------
__global__ __launch_bounds__(256)
void attn_scores(const unsigned short* __restrict__ Qb,
                 const unsigned short* __restrict__ Kb,
                 float* __restrict__ scores)
{
    __shared__ alignas(32) unsigned short Qs[16 * 64];   // q-tile, bf16
    __shared__ float wmax[8][16];
    __shared__ float rowred[16];
    __shared__ float psum[16][16];
    __shared__ float rowinv[16];

    const int tid  = threadIdx.x;
    const int lane = tid & 31;
    const int w    = tid >> 5;
    const int q16  = blockIdx.x & 127;       // S/16
    const int bh   = blockIdx.x >> 7;
    const int b    = bh >> 4, h = bh & 15, kvh = h >> 2;
    const int q0   = q16 * 16;
    const size_t bhS = (size_t)bh * Sc;

    const int m    = lane & 15;
    const int half = lane & 16;

    // Async-stage Q[q0..q0+15][h*64..h*64+63] into LDS: 128 x 16B chunks
    if (tid < 128) {
        const int r = tid >> 3, chunk = tid & 7;         // 8 chunks per row
        const unsigned short* g =
            &Qb[((size_t)(b * Sc) + q0 + r) * 1024 + (size_t)h * 64 + chunk * 8];
        async_copy_b128(lds_off32(&Qs[r * 64 + chunk * 8]), g);
    }
    async_wait0();
    __syncthreads();

    // Q A-fragments (HD=64 -> two 32-wide k-steps) from LDS
    Frag16 aq[2];
    #pragma unroll
    for (int s = 0; s < 2; ++s)
        #pragma unroll
        for (int v = 0; v < 8; ++v)
            aq[s].u[v] = *(const unsigned*)&Qs[m * 64 + a_kbase(v, lane) + 32 * s];

    float rm[8];
    #pragma unroll
    for (int j = 0; j < 8; ++j) rm[j] = NEGF;

    for (int t = 0; t < 16; ++t) {
        const int kt  = t * 8 + w;           // interleave waves over key tiles
        const int k0  = kt * 16;
        const int key = k0 + m;
        const size_t krow = ((size_t)(b * Sc) + key) * 256 + (size_t)kvh * 64;

        v8f acc = {};
        #pragma unroll
        for (int s = 0; s < 2; ++s) {
            Frag16 bk;                       // contiguous 32B v16bf load
            bk.v = *(const v16bf*)&Kb[krow + half + 32 * s];
            WMMA_BF16(acc, aq[s].v, bk.v);
        }
        #pragma unroll
        for (int j = 0; j < 8; ++j) {
            const int qp = q0 + j + (half ? 8 : 0);
            const int kp = k0 + m;
            const float sv = (kp <= qp) ? acc[j] : NEGF;   // causal mask
            scores[(bhS + qp) * (size_t)Sc + kp] = sv;
            rm[j] = fmaxf(rm[j], sv);
        }
    }

    // rowmax reduction: across the 16 lanes of each half (wave32)
    #pragma unroll
    for (int j = 0; j < 8; ++j)
        #pragma unroll
        for (int off = 1; off < 16; off <<= 1)
            rm[j] = fmaxf(rm[j], __shfl_xor(rm[j], off, 32));
    if (m == 0)
        #pragma unroll
        for (int j = 0; j < 8; ++j)
            wmax[w][j + (half ? 8 : 0)] = rm[j];

    __threadfence();
    __syncthreads();
    if (tid < 16) {
        float mm = NEGF;
        for (int w2 = 0; w2 < 8; ++w2) mm = fmaxf(mm, wmax[w2][tid]);
        rowred[tid] = mm;
    }
    __syncthreads();

    // Phase 2: exp(x - max) in place + partial sums (rows stay L2-resident)
    const int row = tid >> 4, seg = tid & 15;
    float* rp = scores + (bhS + q0 + row) * (size_t)Sc + seg * 128;
    const float mx = rowred[row];
    float sum = 0.f;
    #pragma unroll 4
    for (int i = 0; i < 128; ++i) {
        float e = __expf(rp[i] - mx);
        rp[i] = e;
        sum += e;
    }
    psum[row][seg] = sum;
    __threadfence();
    __syncthreads();
    if (tid < 16) {
        float ts = 0.f;
        for (int k = 0; k < 16; ++k) ts += psum[tid][k];
        rowinv[tid] = 1.0f / ts;
    }
    __syncthreads();

    // Phase 3: normalize
    const float inv = rowinv[row];
    #pragma unroll 4
    for (int i = 0; i < 128; ++i) rp[i] *= inv;
}

// ---------------------------------------------------------------------------
// Kernel 3: attn_out = scores @ V (GQA).  Block = (b,h, 64 q-rows), 4 waves.
// V tile transposed in LDS so B-fragments are contiguous v16bf LDS reads.
// (Transpose-on-store requires the VALU path; async copy can't reshape.)
// ---------------------------------------------------------------------------
__global__ __launch_bounds__(128)
void attn_pv(const float* __restrict__ scores,
             const unsigned short* __restrict__ Vb,
             unsigned short* __restrict__ Ab)
{
    __shared__ alignas(32) unsigned short Vt[64 * 32];   // [hd][key]

    const int tid  = threadIdx.x;
    const int lane = tid & 31;
    const int w    = tid >> 5;               // 0..3
    const int qt   = blockIdx.x & 31;        // S/64
    const int bh   = blockIdx.x >> 5;
    const int b    = bh >> 4, h = bh & 15, kvh = h >> 2;
    const int q0   = qt * 64 + w * 16;
    const size_t bhS = (size_t)bh * Sc;

    const int m    = lane & 15;
    const int half = lane & 16;
    const float* srow = scores + (bhS + q0 + m) * (size_t)Sc;

    v8f acc[4] = {};

    for (int kk = 0; kk < Sc; kk += 32) {
        __syncthreads();
        // stage V[kk..kk+31][0..63] transposed into LDS
        #pragma unroll
        for (int i = 0; i < 16; ++i) {
            int e = tid + 128 * i;
            int key = e >> 6, hd = e & 63;
            Vt[hd * 32 + key] =
                Vb[((size_t)(b * Sc) + kk + key) * 256 + (size_t)kvh * 64 + hd];
        }
        __syncthreads();

        Frag16 a;                            // scores fp32 -> bf16 on the fly
        #pragma unroll
        for (int v = 0; v < 8; ++v) {
            float2 p = *(const float2*)&srow[kk + a_kbase(v, lane)];
            a.us[2 * v]     = f2bf(p.x);
            a.us[2 * v + 1] = f2bf(p.y);
        }
        #pragma unroll
        for (int c = 0; c < 4; ++c) {
            Frag16 bv;
            bv.v = *(const v16bf*)&Vt[(c * 16 + m) * 32 + half];
            WMMA_BF16(acc[c], a.v, bv.v);
        }
    }

    #pragma unroll
    for (int c = 0; c < 4; ++c)
        #pragma unroll
        for (int j = 0; j < 8; ++j) {
            const int qp = q0 + j + (half ? 8 : 0);
            Ab[((size_t)(b * Sc) + qp) * 1024 + (size_t)h * 64 + c * 16 + m] =
                f2bf(acc[c][j]);
        }
}

// ---------------------------------------------------------------------------
// Kernel 4: out = attn @ o_w^T + o_b  (4096x1024 @ 1024x1024), fp32 output.
// A tile is already bf16 -> async global->LDS copy (ASYNCcnt path).
// ---------------------------------------------------------------------------
__global__ __launch_bounds__(256)
void o_proj(const unsigned short* __restrict__ Ab, const float* __restrict__ Wo,
            const float* __restrict__ bo, float* __restrict__ out)
{
    __shared__ alignas(32) unsigned short As[128 * 32];
    __shared__ alignas(32) unsigned short Bs[64 * 32];

    const int tid  = threadIdx.x;
    const int lane = tid & 31;
    const int w    = tid >> 5;
    const int ntile = blockIdx.x & 15;       // 1024 / 64
    const int mtile = blockIdx.x >> 4;       // 4096 / 128
    const int m0 = mtile * 128, n0 = ntile * 64;
    const int m    = lane & 15;
    const int half = lane & 16;

    v8f acc[4] = {};

    for (int kk = 0; kk < 1024; kk += 32) {
        __syncthreads();
        // stage A via async global->LDS (pure bf16 copy): 512 x 16B chunks
        #pragma unroll
        for (int i = 0; i < 2; ++i) {
            int e = tid + 256 * i;           // 0..511
            int r = e >> 2, chunk = e & 3;   // 128 rows x 4 chunks of 8 ushorts
            async_copy_b128(lds_off32(&As[r * 32 + chunk * 8]),
                            &Ab[(size_t)(m0 + r) * 1024 + kk + chunk * 8]);
        }
        // stage B (fp32 -> bf16, VALU path)
        #pragma unroll
        for (int i = 0; i < 8; ++i) {
            int e = tid + 256 * i;
            int r = e >> 5, c = e & 31;
            Bs[e] = f2bf(Wo[(size_t)(n0 + r) * 1024 + kk + c]);
        }
        if (kk + 32 < 1024)
            __builtin_prefetch(&Wo[(size_t)(n0 + (tid & 63)) * 1024 + kk + 32], 0, 1);
        async_wait0();
        __syncthreads();

        Frag16 a;
        #pragma unroll
        for (int v = 0; v < 8; ++v)
            a.u[v] = *(const unsigned*)&As[(16 * w + m) * 32 + a_kbase(v, lane)];
        #pragma unroll
        for (int c = 0; c < 4; ++c) {
            Frag16 b;
            b.v = *(const v16bf*)&Bs[(c * 16 + m) * 32 + half];
            WMMA_BF16(acc[c], a.v, b.v);
        }
    }

    const int mrow = m0 + 16 * w + (half ? 8 : 0);
    #pragma unroll
    for (int c = 0; c < 4; ++c) {
        const int gn = n0 + c * 16 + m;
        const float bb = bo[gn];
        #pragma unroll
        for (int j = 0; j < 8; ++j)
            out[(size_t)(mrow + j) * 1024 + gn] = acc[c][j] + bb;
    }
}

// ---------------------------------------------------------------------------
extern "C" void kernel_launch(void* const* d_in, const int* in_sizes, int n_in,
                              void* d_out, int out_size, void* d_ws, size_t ws_size,
                              hipStream_t stream) {
    const float* hs   = (const float*)d_in[0];   // hidden_states (B,S,H)
    // d_in[1] = mask — causal mask is computed analytically in-kernel
    const float* scal = (const float*)d_in[2];   // scaling (HD)
    const float* qkvw = (const float*)d_in[3];   // (1536, 1024)
    const float* qkvb = (const float*)d_in[4];   // (1536)
    const float* ow   = (const float*)d_in[5];   // (1024, 1024)
    const float* ob   = (const float*)d_in[6];   // (1024)

    // Workspace: bf16 intermediates (total ~20 MB)
    unsigned short* Qb = (unsigned short*)d_ws;            // NTOK x 1024
    unsigned short* Kb = Qb + (size_t)NTOK * 1024;         // NTOK x 256
    unsigned short* Vb = Kb + (size_t)NTOK * 256;          // NTOK x 256
    unsigned short* Ab = Vb + (size_t)NTOK * 256;          // NTOK x 1024

    float* scores = (float*)d_out;                               // B*NH*S*S
    float* out    = scores + (size_t)Bc * NHc * Sc * Sc;         // B*S*1024

    qkv_gemm   <<<dim3(32 * 24), 256, 0, stream>>>(hs, qkvw, qkvb, scal, Qb, Kb, Vb);
    attn_scores<<<dim3(Bc * NHc * (Sc / 16)), 256, 0, stream>>>(Qb, Kb, scores);
    attn_pv    <<<dim3(Bc * NHc * (Sc / 64)), 128, 0, stream>>>(scores, Vb, Ab);
    o_proj     <<<dim3(32 * 16), 256, 0, stream>>>(Ab, ow, ob, out);
}
